// HPUAttentionImpl_23699629539461
// MI455X (gfx1250) — matrix operand order repaired and checked by reading, not compiled, not run on hardware
//
#include <hip/hip_runtime.h>

#define BATCH 32
#define NHEADS 32
#define KVH 8
#define QPK 4
#define HD 128
#define BS 128
#define BPS 16
#define NW 8
#define SCALEQ 0.088388347648318447f
#define NEGV -1.0e30f

typedef __attribute__((ext_vector_type(16))) __bf16 v16bf;
typedef __attribute__((ext_vector_type(8)))  __bf16 v8bf;
typedef __attribute__((ext_vector_type(4)))  __bf16 v4bf;
typedef __attribute__((ext_vector_type(8)))  float  v8f;
typedef __attribute__((ext_vector_type(4)))  float  v4f;

// Native bf16 conversion: vector fptrunc lowers to v_cvt_pk_bf16_f32 on gfx1250.
static __device__ __forceinline__ v16bf pack16(v4f a, v4f b, v4f c, v4f d) {
  v4bf x = __builtin_convertvector(a, v4bf);
  v4bf y = __builtin_convertvector(b, v4bf);
  v4bf z = __builtin_convertvector(c, v4bf);
  v4bf w = __builtin_convertvector(d, v4bf);
  v8bf lo = __builtin_shufflevector(x, y, 0, 1, 2, 3, 4, 5, 6, 7);
  v8bf hi = __builtin_shufflevector(z, w, 0, 1, 2, 3, 4, 5, 6, 7);
  return __builtin_shufflevector(lo, hi, 0, 1, 2, 3, 4, 5, 6, 7, 8, 9, 10, 11,
                                 12, 13, 14, 15);
}

static __device__ __forceinline__ v8f wmma_bf16(v16bf a, v16bf b, v8f c) {
  // D = A(16x32 bf16) * B(32x16 bf16) + C(16x16 f32)
  return __builtin_amdgcn_wmma_f32_16x16x32_bf16(false, a, false, b, (short)0, c,
                                                 false, false);
}

// Static LDS: ~59 KB per workgroup -> 2 workgroups (16 waves) per WGP
struct SMem {
  float  stage[NW][32 * 32];   // per-wave f32 V quarter [32 tok][32 d]   (4 KB)
  __bf16 vt[NW][32 * 32];      // per-wave bf16 V^T quarter [32 d][32 tok] (2 KB)
  __bf16 el[NW][QPK * BS];     // per-wave bf16 probs rows 0..3            (1 KB)
  __bf16 zp[BS];               // shared zero page = A rows 4..15
  float  outl[QPK * HD];
  float  mlds[BPS * QPK];
  float  llds[BPS * QPK];
};

__global__ __launch_bounds__(NW * 32) void
pa_alibi_flash(const float* __restrict__ q,
               const float* __restrict__ kcache,
               const float* __restrict__ vcache,
               const float* __restrict__ ab,
               const float* __restrict__ slopes,
               const int* __restrict__ blist,
               const int* __restrict__ busage,
               float* __restrict__ out)
{
  __shared__ SMem sm;

  const int g    = blockIdx.x;   // kv head
  const int b    = blockIdx.y;   // sequence
  const int tid  = threadIdx.x;
  const int wave = tid >> 5;
  const int lane = tid & 31;
  const int lr   = lane & 15;    // position within lane half
  const int hif  = lane >> 4;    // lane half

  for (int i = tid; i < QPK * HD; i += NW * 32) sm.outl[i] = 0.f;
  if (wave == 0) {
    uint4 z; z.x = 0u; z.y = 0u; z.z = 0u; z.w = 0u;
    for (int i = lane; i < (int)(BS * 2 / 16); i += 32) ((uint4*)sm.zp)[i] = z;
  }
  __syncthreads();   // zero page + outl visible to everyone

  float*  stg = sm.stage[wave];
  __bf16* vt  = sm.vt[wave];
  __bf16* el  = sm.el[wave];

  float sl[QPK];
#pragma unroll
  for (int h = 0; h < QPK; ++h) sl[h] = slopes[g * QPK + h];

  const v4f z4 = {0.f, 0.f, 0.f, 0.f};
  const v8f z8 = {0.f, 0.f, 0.f, 0.f, 0.f, 0.f, 0.f, 0.f};

  // ---- Q A-fragments: rows 0..3 = query heads of this kv head, rows 4..15 zero ----
  v16bf aq[4];
  {
    const int  m    = lr;
    const bool vrow = (m < QPK);
    const float* qp = q + ((size_t)b * NHEADS + (size_t)g * QPK + (vrow ? m : 0)) * HD;
#pragma unroll
    for (int c = 0; c < 4; ++c) {
      const int b0 = c * 32 + (hif ? 8 : 0);
      const int b1 = c * 32 + (hif ? 24 : 16);
      v4f q0 = z4, q1 = z4, q2 = z4, q3 = z4;
      if (vrow) {
        q0 = *(const v4f*)(qp + b0);
        q1 = *(const v4f*)(qp + b0 + 4);
        q2 = *(const v4f*)(qp + b1);
        q3 = *(const v4f*)(qp + b1 + 4);
      }
      aq[c] = pack16(q0 * SCALEQ, q1 * SCALEQ, q2 * SCALEQ, q3 * SCALEQ);
    }
  }

  v8f acc[2][8];
#pragma unroll
  for (int rp = 0; rp < 2; ++rp)
#pragma unroll
    for (int n = 0; n < 8; ++n) acc[rp][n] = z8;

  float msave[2][QPK];

  for (int rp = 0; rp < 2; ++rp) {
    const int slot  = wave + rp * NW;       // block index within the sequence
    const int u     = b * BPS + slot;
    const int blk   = blist[u];
    const int usage = busage[u];

    // L2 prefetch of next block's K/V stream (global_prefetch_b8)
    if (rp == 0) {
      const int blk2 = blist[u + NW];
      __builtin_prefetch(kcache + (((size_t)blk2 * BS + lane * 4) * KVH + g) * HD, 0, 1);
      __builtin_prefetch(vcache + (((size_t)blk2 * BS + lane * 4) * KVH + g) * HD, 0, 1);
    }

    // ---------------- QK^T ----------------
    v8f s[8];
#pragma unroll
    for (int t = 0; t < 8; ++t) {
      s[t] = z8;
#pragma unroll
      for (int c = 0; c < 4; ++c) {
        // B fragment: lane = token column, elements = d in table order (K rows)
        const float* kp = kcache +
            (((size_t)blk * BS + (size_t)t * 16 + lr) * KVH + g) * HD + c * 32 +
            (hif ? 8 : 0);
        v4f k0 = *(const v4f*)(kp);
        v4f k1 = *(const v4f*)(kp + 4);
        v4f k2 = *(const v4f*)(kp + 16);
        v4f k3 = *(const v4f*)(kp + 20);
        s[t] = wmma_bf16(aq[c], pack16(k0, k1, k2, k3), s[t]);
      }
    }

    // ------------- alibi + masks (valid head rows only: low half, r<QPK) -------------
#pragma unroll
    for (int t = 0; t < 8; ++t) {
      const int   tok   = t * 16 + lr;
      const float abv   = ab[(size_t)u * BS + tok];
      const bool  tokOK = tok < usage;
#pragma unroll
      for (int r = 0; r < QPK; ++r) {
        float v = s[t][r];
        if (hif == 0 && tokOK) v += abv * sl[r];
        else v = NEGV;
        s[t][r] = v;
      }
    }

    // ---------------- per-block softmax stats ----------------
    float mr[QPK];
#pragma unroll
    for (int r = 0; r < QPK; ++r) {
      float mv = NEGV;
#pragma unroll
      for (int t = 0; t < 8; ++t) mv = fmaxf(mv, s[t][r]);
#pragma unroll
      for (int o = 1; o < 16; o <<= 1) mv = fmaxf(mv, __shfl_xor(mv, o, 32));
      mr[r] = mv;
      msave[rp][r] = mv;
    }
#pragma unroll
    for (int t = 0; t < 8; ++t) {
      const int  tok   = t * 16 + lr;
      const bool tokOK = tok < usage;
#pragma unroll
      for (int r = 0; r < QPK; ++r) {
        float e = 0.f;
        if (hif == 0 && tokOK) e = __expf(s[t][r] - mr[r]);
        s[t][r] = e;   // reuse tiles for probabilities
      }
    }
    float ls[QPK];
#pragma unroll
    for (int r = 0; r < QPK; ++r) {
      float sum = 0.f;
#pragma unroll
      for (int t = 0; t < 8; ++t) sum += s[t][r];
#pragma unroll
      for (int o = 1; o < 16; o <<= 1) sum += __shfl_xor(sum, o, 32);
      ls[r] = sum;
    }
    if (lane == 0) {
#pragma unroll
      for (int h = 0; h < QPK; ++h) {
        sm.mlds[slot * QPK + h] = mr[h];
        sm.llds[slot * QPK + h] = ls[h];
      }
    }

    // probabilities -> LDS bf16 (A-matrix rows 0..3; rows 4..15 = shared zero page)
    if (hif == 0) {
#pragma unroll
      for (int t = 0; t < 8; ++t)
#pragma unroll
        for (int r = 0; r < QPK; ++r)
          el[r * BS + t * 16 + lr] = (__bf16)s[t][r];
    }

    // ---------------- AV ----------------
    for (int kc = 0; kc < 4; ++kc) {
      // A fragment of probabilities for this 32-token chunk
      const __bf16* arow = (lr < QPK) ? (el + lr * BS + kc * 32) : sm.zp;
      v8bf alo = *(const v8bf*)(arow + (hif ? 8 : 0));
      v8bf ahi = *(const v8bf*)(arow + (hif ? 24 : 16));
      v16bf aE = __builtin_shufflevector(alo, ahi, 0, 1, 2, 3, 4, 5, 6, 7, 8, 9,
                                         10, 11, 12, 13, 14, 15);

      const float* vbase =
          vcache + (((size_t)blk * BS + (size_t)kc * 32) * KVH + g) * HD;

      for (int dq = 0; dq < 4; ++dq) {   // 32-wide d quarters
        // stage buffer reuse: wave ds reads of the previous quarter must land
        // before the async engine overwrites it (async vs DS are unordered)
        asm volatile("s_wait_dscnt 0x0" ::: "memory");

        // async-stage f32 V quarter [32 tok][32 d] into per-wave LDS
        for (int j = 0; j < 8; ++j) {
          const int fi  = j * 32 + lane;   // 16-byte units
          const int row = fi >> 3;         // token 0..31
          const int cu  = fi & 7;          // 16B column unit
          const float* gp = vbase + (size_t)row * KVH * HD + dq * 32 + cu * 4;
          unsigned ldsoff =
              (unsigned)((const char*)(stg + row * 32 + cu * 4) - (const char*)&sm);
          unsigned long long ga = (unsigned long long)gp;
          asm volatile("global_load_async_to_lds_b128 %0, %1, off"
                       :: "v"(ldsoff), "v"(ga) : "memory");
        }
        asm volatile("s_wait_asynccnt 0x0" ::: "memory");

        // convert + transpose into vt[d][tok] bf16 (lane = its own token row)
        for (int j = 0; j < 8; ++j) {
          v4f vv = *(const v4f*)(stg + lane * 32 + j * 4);
          v4bf cb = __builtin_convertvector(vv, v4bf);
#pragma unroll
          for (int i = 0; i < 4; ++i) vt[(j * 4 + i) * 32 + lane] = cb[i];
        }

        // two 16-wide d tiles of WMMA for this quarter
#pragma unroll
        for (int nn = 0; nn < 2; ++nn) {
          const int n = dq * 2 + nn;
          const __bf16* vp = vt + (nn * 16 + lr) * 32 + (hif ? 8 : 0);
          v8bf b0 = *(const v8bf*)(vp);
          v8bf b1 = *(const v8bf*)(vp + 16);
          v16bf bV = __builtin_shufflevector(b0, b1, 0, 1, 2, 3, 4, 5, 6, 7, 8,
                                             9, 10, 11, 12, 13, 14, 15);
          acc[rp][n] = wmma_bf16(aE, bV, acc[rp][n]);
        }
      }
    }
  }

  __syncthreads();

  // global (per-sequence) softmax stats from all 16 blocks
  float mg[QPK], lg[QPK];
#pragma unroll
  for (int h = 0; h < QPK; ++h) {
    float mv = NEGV;
    for (int sct = 0; sct < BPS; ++sct) mv = fmaxf(mv, sm.mlds[sct * QPK + h]);
    mg[h] = mv;
    float sum = 0.f;
    for (int sct = 0; sct < BPS; ++sct)
      sum += __expf(sm.mlds[sct * QPK + h] - mv) * sm.llds[sct * QPK + h];
    lg[h] = sum;
  }

  // rescale each wave's partial AV and reduce into out_lds
#pragma unroll
  for (int rp = 0; rp < 2; ++rp) {
    if (hif == 0) {
#pragma unroll
      for (int r = 0; r < QPK; ++r) {
        const float sc = __expf(msave[rp][r] - mg[r]);
#pragma unroll
        for (int n = 0; n < 8; ++n)
          atomicAdd(&sm.outl[r * HD + n * 16 + lr], sc * acc[rp][n][r]);
      }
    }
  }

  __syncthreads();

  for (int i = tid; i < QPK * HD; i += NW * 32) {
    const int h = i >> 7;
    const int d = i & (HD - 1);
    out[((size_t)b * NHEADS + (size_t)g * QPK + h) * HD + d] = sm.outl[i] / lg[h];
  }
}

extern "C" void kernel_launch(void* const* d_in, const int* in_sizes, int n_in,
                              void* d_out, int out_size, void* d_ws, size_t ws_size,
                              hipStream_t stream) {
  (void)in_sizes; (void)n_in; (void)out_size; (void)d_ws; (void)ws_size;
  const float* q      = (const float*)d_in[0];
  const float* kcache = (const float*)d_in[1];
  const float* vcache = (const float*)d_in[2];
  const float* ab     = (const float*)d_in[3];
  const float* slopes = (const float*)d_in[4];
  const int*   blist  = (const int*)d_in[5];
  // d_in[6] = block_groups: implied by contiguous layout (u / BPS)
  const int*   busage = (const int*)d_in[7];
  float* out = (float*)d_out;

  dim3 grid(KVH, BATCH);   // (kv head, sequence)
  dim3 blk(NW * 32);
  pa_alibi_flash<<<grid, blk, 0, stream>>>(q, kcache, vcache, ab, slopes, blist,
                                           busage, out);
}
